// DirectionLoss_43344809951441
// MI455X (gfx1250) — compile-verified, matrix operand order checked by reading, not run on hardware
//
#include <hip/hip_runtime.h>
#include <hip/hip_bf16.h>

typedef __attribute__((ext_vector_type(16))) _Float16 v16h;
typedef __attribute__((ext_vector_type(8)))  float    v8f;

#if defined(__has_builtin)
#if __has_builtin(__builtin_amdgcn_tensor_load_to_lds) && __has_builtin(__builtin_amdgcn_s_wait_tensorcnt)
#define HAVE_TDM 1
#endif
#endif

#ifdef HAVE_TDM
typedef unsigned int u32x4 __attribute__((ext_vector_type(4)));
typedef int          i32x4 __attribute__((ext_vector_type(4)));
typedef int          i32x8 __attribute__((ext_vector_type(8)));
#endif

#define NN      4
#define HH      1024
#define WW      1024
#define HWSZ    (HH * WW)
#define MAXIT   128
#define TBX     64
#define TBY     4

// flags[0] = active, flags[1] = changed-this-iteration, flags[2] = block counter
__global__ void k_init(int* flags) { flags[0] = 1; flags[1] = 0; flags[2] = 0; }

// binary mask: softmax(logits)[:,1] >= 0.5  <=>  l1 >= l0
__global__ void k_mask(const float* __restrict__ logits,
                       unsigned char* __restrict__ mask, int nhw) {
  int p = blockIdx.x * blockDim.x + threadIdx.x;
  if (p >= nhw) return;
  int n = p / HWSZ, q = p - n * HWSZ;
  float l0 = logits[(size_t)(2 * n)     * HWSZ + q];
  float l1 = logits[(size_t)(2 * n + 1) * HWSZ + q];
  mask[p] = (l1 >= l0) ? 1 : 0;
}

#ifdef HAVE_TDM
// One TDM descriptor: DMA a tileW x tileH byte tile (row stride = rowStride)
// from global gsrc into LDS at lds_off. 2D -> groups 2/3 zero (NULL-equivalent).
__device__ __forceinline__ void tdm_tile_load(unsigned lds_off,
                                              const unsigned char* gsrc,
                                              unsigned remW, unsigned remH,
                                              unsigned tileW, unsigned tileH,
                                              unsigned rowStride) {
  unsigned long long ga = (unsigned long long)(size_t)gsrc;
  u32x4 g0;
  g0.x = 1u;                                       // count=1, user-mode D#
  g0.y = lds_off;                                  // lds_addr [63:32]
  g0.z = (unsigned)(ga & 0xffffffffu);             // global_addr [95:64]
  g0.w = (unsigned)((ga >> 32) & 0x01ffffffu)      // global_addr [120:96]
       | (2u << 30);                               // type = 2 ("image")
  i32x8 g1;
  g1[0] = 0;                                       // wg_mask=0, data_size=1B
  g1[1] = (int)((remW & 0xffffu) << 16);           // tensor_dim0 lo16 @ [63:48]
  g1[2] = (int)(((remW >> 16) & 0xffffu)           // tensor_dim0 hi16 @ [79:64]
       |        ((remH & 0xffffu) << 16));         // tensor_dim1 lo16 @ [95:80]
  g1[3] = (int)(((remH >> 16) & 0xffffu)           // tensor_dim1 hi16 @ [111:96]
       |        ((tileW & 0xffffu) << 16));        // tile_dim0 @ [127:112]
  g1[4] = (int)(tileH & 0xffffu);                  // tile_dim1 @ [143:128], tile_dim2=0
  g1[5] = (int)rowStride;                          // tensor_dim0_stride lo32 @ [191:160]
  g1[6] = 0;                                       // stride hi16 = 0, dim1_stride = 0
  g1[7] = 0;
  i32x4 g2 = {}, g3 = {};
#if __clang_major__ >= 23
  i32x8 g4 = {};
  __builtin_amdgcn_tensor_load_to_lds(g0, g1, g2, g3, g4, 0);
#else
  __builtin_amdgcn_tensor_load_to_lds(g0, g1, g2, g3, 0);
#endif
  __builtin_amdgcn_s_wait_tensorcnt(0);
}
#endif

// one Zhang-Suen substep, LDS-tiled 3x3 stencil, gated by device-side flag.
// sub==1 additionally performs the convergence-flag update in the last block.
__global__ void k_thin(const unsigned char* __restrict__ src,
                       unsigned char* __restrict__ dst, int sub,
                       int* __restrict__ flags, int gridBlocks) {
  if (flags[0] == 0) return;   // converged: one scalar load, no work
  __shared__ unsigned char t[(TBY + 2) * (TBX + 2)];
  const int n  = blockIdx.z;
  const int x0 = blockIdx.x * TBX, y0 = blockIdx.y * TBY;
  const unsigned char* s = src + (size_t)n * HWSZ;
  const int TW = TBX + 2;
  const int tid = threadIdx.y * TBX + threadIdx.x;

#ifdef HAVE_TDM
  const bool interior = (x0 > 0) && (y0 > 0) &&
                        (x0 + TBX + 1 < WW) && (y0 + TBY + 1 < HH);
  if (interior) {
    if (tid < 32) {   // wave 0 issues the DMA and waits TENSORcnt
      unsigned lds_off =
          (unsigned)(unsigned long long)(__attribute__((address_space(3))) unsigned char*)t;
      const unsigned char* g = s + (size_t)(y0 - 1) * WW + (x0 - 1);
      tdm_tile_load(lds_off, g, (unsigned)(WW - (x0 - 1)), (unsigned)(HH - (y0 - 1)),
                    (unsigned)TW, (unsigned)(TBY + 2), (unsigned)WW);
    }
  } else
#endif
  {
    // manual halo load (borders / no-TDM toolchains)
    {
      int py = y0 + TBY; if (py >= HH) py = HH - 1;
      __builtin_prefetch(s + (size_t)py * WW + x0, 0, 1);
    }
    for (int idx = tid; idx < (TBY + 2) * TW; idx += TBX * TBY) {
      int ty = idx / TW, tx = idx - ty * TW;
      int gy = y0 - 1 + ty, gx = x0 - 1 + tx;
      unsigned char v = 0;
      if (gy >= 0 && gy < HH && gx >= 0 && gx < WW) v = s[(size_t)gy * WW + gx];
      t[idx] = v;
    }
  }
  __syncthreads();

  int lx = threadIdx.x, ly = threadIdx.y;
  int x = x0 + lx, y = y0 + ly;
  if (x < WW && y < HH) {
    int tx = lx + 1, ty = ly + 1;
    int P1 = t[ty * TW + tx];
    int P2 = t[(ty - 1) * TW + tx    ];
    int P3 = t[(ty - 1) * TW + tx + 1];
    int P4 = t[ ty      * TW + tx + 1];
    int P5 = t[(ty + 1) * TW + tx + 1];
    int P6 = t[(ty + 1) * TW + tx    ];
    int P7 = t[(ty + 1) * TW + tx - 1];
    int P8 = t[ ty      * TW + tx - 1];
    int P9 = t[(ty - 1) * TW + tx - 1];
    int A = 0;
    A += (P2 == 0) & (P3 == 1);  A += (P3 == 0) & (P4 == 1);
    A += (P4 == 0) & (P5 == 1);  A += (P5 == 0) & (P6 == 1);
    A += (P6 == 0) & (P7 == 1);  A += (P7 == 0) & (P8 == 1);
    A += (P8 == 0) & (P9 == 1);  A += (P9 == 0) & (P2 == 1);
    int Bc = P2 + P3 + P4 + P5 + P6 + P7 + P8 + P9;
    bool cond = (P1 == 1) && (Bc >= 2) && (Bc <= 6) && (A == 1);
    if (sub == 0) cond = cond && (P2 * P4 * P6 == 0) && (P4 * P6 * P8 == 0);
    else          cond = cond && (P2 * P4 * P8 == 0) && (P2 * P6 * P8 == 0);
    unsigned char nv = cond ? 0 : (unsigned char)P1;
    dst[(size_t)n * HWSZ + (size_t)y * WW + x] = nv;
    if (cond) atomicOr(&flags[1], 1);   // 1->0 only: any substep change == iter change
  }

  if (sub == 1) {
    __syncthreads();                    // all block threads reached here
    if (tid == 0) {
      __threadfence();
      int prev = atomicAdd(&flags[2], 1);
      if (prev == gridBlocks - 1) {     // last block: fold former k_update
        __threadfence();
        int ch = atomicExch(&flags[1], 0);
        flags[0] = (ch != 0) ? 1 : 0;
        atomicExch(&flags[2], 0);
      }
    }
  }
}

// conv-1 for all 4 directions via one v_wmma_f32_16x16x32_f16 per wave.
// A: 16 pixels x 25-tap 5x5 im2col (K padded to 32); B cols 0..3 = dir kernels.
// C_d = (conv == 2). Writes 4 planes of uint8.
__global__ void k_cmap(const unsigned char* __restrict__ Bs,
                       unsigned char* __restrict__ C) {
  __shared__ unsigned char tile[12 * 20];          // (8+4) x (16+4) halo tile
  const int n  = blockIdx.z;
  const int x0 = blockIdx.x * 16, y0 = blockIdx.y * 8;
  const unsigned char* s = Bs + (size_t)n * HWSZ;
  int tid = threadIdx.x;
  if (tid < 240) {
    int ty = tid / 20, tx = tid - ty * 20;
    int gy = y0 - 2 + ty, gx = x0 - 2 + tx;
    unsigned char v = 0;
    if (gy >= 0 && gy < HH && gx >= 0 && gx < WW) v = s[(size_t)gy * WW + gx];
    tile[tid] = v;
  }
  __syncthreads();
  int lane = tid & 31, w = tid >> 5;   // wave w -> row y0+w
  int m  = lane & 15;                  // M (pixel) for A, N (direction) for B
  int lo = (lane < 16) ? 0 : 8, hi = (lane < 16) ? 16 : 24;
  v16h a, b;
#pragma unroll
  for (int h = 0; h < 16; ++h) {
    int K = (h < 8) ? (lo + h) : (hi + h - 8);   // ISA 7.12.2 16-bit A layout
    float av = 0.f, bv = 0.f;
    if (K < 25) {
      int i = K / 5, j = K - 5 * i;              // tap (y-2+i, x-2+j)
      av = (float)tile[(w + i) * 20 + (m + j)];
      if      (m == 0) bv = (i == 2)     ? 1.f : 0.f;   // horizontal 1x5
      else if (m == 1) bv = (j == 2)     ? 1.f : 0.f;   // vertical 5x1
      else if (m == 2) bv = (j == 4 - i) ? 1.f : 0.f;   // anti-diagonal (k_d1)
      else if (m == 3) bv = (i == j)     ? 1.f : 0.f;   // diagonal (k_b1)
    }
    a[h] = (_Float16)av;
    b[h] = (_Float16)bv;
  }
  v8f c = {};
  c = __builtin_amdgcn_wmma_f32_16x16x32_f16(false, a, false, b, (short)0, c, false, false);
  if (m < 4) {
    int M0 = (lane < 16) ? 0 : 8;
    int y = y0 + w;
    size_t base = (size_t)m * ((size_t)NN * HWSZ) + (size_t)n * HWSZ
                + (size_t)y * WW + x0 + M0;
#pragma unroll
    for (int r = 0; r < 8; ++r)
      C[base + r] = (c[r] == 2.0f) ? 1 : 0;
  }
}

// conv-2 (x10) for all 4 directions via one WMMA; K slot = d*8 + tap.
__global__ void k_dmap(const unsigned char* __restrict__ C,
                       unsigned char* __restrict__ Dm) {
  __shared__ unsigned char tile[4][12 * 20];
  const int n  = blockIdx.z;
  const int x0 = blockIdx.x * 16, y0 = blockIdx.y * 8;
  for (int idx = threadIdx.x; idx < 4 * 240; idx += 256) {
    int d = idx / 240, r = idx - d * 240;
    int ty = r / 20, tx = r - ty * 20;
    int gy = y0 - 2 + ty, gx = x0 - 2 + tx;
    unsigned char v = 0;
    if (gy >= 0 && gy < HH && gx >= 0 && gx < WW)
      v = C[(size_t)d * ((size_t)NN * HWSZ) + (size_t)n * HWSZ + (size_t)gy * WW + gx];
    tile[d][r] = v;
  }
  __syncthreads();
  int tid = threadIdx.x, lane = tid & 31, w = tid >> 5;
  int m  = lane & 15;
  int lo = (lane < 16) ? 0 : 8, hi = (lane < 16) ? 16 : 24;
  v16h a, b;
#pragma unroll
  for (int h = 0; h < 16; ++h) {
    int K = (h < 8) ? (lo + h) : (hi + h - 8);
    int d = K >> 3, tp = K & 7;
    float av = 0.f, bv = 0.f;
    if (tp < 5) {
      int ty, tx;
      if      (d == 0) { ty = w + 2;  tx = m + tp;     }  // horizontal
      else if (d == 1) { ty = w + tp; tx = m + 2;      }  // vertical
      else if (d == 2) { ty = w + tp; tx = m + 4 - tp; }  // anti-diagonal
      else             { ty = w + tp; tx = m + tp;     }  // diagonal
      av = (float)tile[d][ty * 20 + tx];
      if (m == d) bv = 10.f;                              // k1 * 10
    }
    a[h] = (_Float16)av;
    b[h] = (_Float16)bv;
  }
  v8f c = {};
  c = __builtin_amdgcn_wmma_f32_16x16x32_f16(false, a, false, b, (short)0, c, false, false);
  int M0 = (lane < 16) ? 0 : 8;
#pragma unroll
  for (int r = 0; r < 8; ++r) {
    float dd = c[r];
    dd = fminf(dd, 10.f);
    dd = (dd == 0.f) ? 1.f : dd;
    dd += __shfl_xor(dd, 1, 32);    // sum directions across lanes 0..3 / 16..19
    dd += __shfl_xor(dd, 2, 32);
    float D = (dd >= 10.f) ? 10.f : dd;
    D = (D == 2.f) ? 1.f : D;
    if (m == 0) {
      int y = y0 + w, x = x0 + M0 + r;
      Dm[(size_t)n * HWSZ + (size_t)y * WW + x] = (unsigned char)D;
    }
  }
}

// per-block fixed-order partial sums of D * ce  (no float atomics -> deterministic)
__global__ void k_reduce(const float* __restrict__ logits, const int* __restrict__ target,
                         const unsigned char* __restrict__ Dm,
                         float* __restrict__ partials, int nhw) {
  __shared__ float sm[256];
  int p = blockIdx.x * 256 + threadIdx.x;
  float v = 0.f;
  if (p < nhw) {
    int n = p / HWSZ, q = p - n * HWSZ;
    float l0 = logits[(size_t)(2 * n)     * HWSZ + q];
    float l1 = logits[(size_t)(2 * n + 1) * HWSZ + q];
    float mx  = fmaxf(l0, l1);
    float lse = mx + __logf(__expf(l0 - mx) + __expf(l1 - mx));
    float lt  = target[p] ? l1 : l0;
    v = (float)Dm[p] * (lse - lt);
  }
  sm[threadIdx.x] = v;
  __syncthreads();
  for (int s = 128; s > 0; s >>= 1) {
    if (threadIdx.x < s) sm[threadIdx.x] += sm[threadIdx.x + s];
    __syncthreads();
  }
  if (threadIdx.x == 0) partials[blockIdx.x] = sm[0];
}

__global__ void k_final(const float* __restrict__ partials, float* __restrict__ out,
                        int blocksPerImg) {
  __shared__ float sm[256];
  float acc = 0.f;
  for (int n = 0; n < NN; ++n) {
    float local = 0.f;
    for (int i = threadIdx.x; i < blocksPerImg; i += 256)
      local += partials[n * blocksPerImg + i];
    sm[threadIdx.x] = local;
    __syncthreads();
    for (int s = 128; s > 0; s >>= 1) {
      if (threadIdx.x < s) sm[threadIdx.x] += sm[threadIdx.x + s];
      __syncthreads();
    }
    if (threadIdx.x == 0) acc += sm[0] / (float)HWSZ;
    __syncthreads();
  }
  if (threadIdx.x == 0) out[0] = acc / (float)NN * 1.0f;   // K_SCALE = 1.0
}

extern "C" void kernel_launch(void* const* d_in, const int* in_sizes, int n_in,
                              void* d_out, int out_size, void* d_ws, size_t ws_size,
                              hipStream_t stream) {
  const float* logits = (const float*)d_in[0];
  const int*   target = (const int*)d_in[1];
  float*       out    = (float*)d_out;

  const int nhw = NN * HWSZ;
  // workspace layout (~29.4 MB): masks x2, 4 C-planes, D-plane, partials, flags
  unsigned char* bufA = (unsigned char*)d_ws;
  unsigned char* bufB = bufA + (size_t)nhw;
  unsigned char* Cpl  = bufB + (size_t)nhw;
  unsigned char* Dm   = Cpl  + 4 * (size_t)nhw;
  float* partials     = (float*)(Dm + (size_t)nhw);
  int*   flags        = (int*)(partials + nhw / 256);

  k_init<<<1, 1, 0, stream>>>(flags);
  k_mask<<<nhw / 256, 256, 0, stream>>>(logits, bufA, nhw);

  dim3 tg(WW / TBX, HH / TBY, NN), tb(TBX, TBY);
  const int gridBlocks = (WW / TBX) * (HH / TBY) * NN;
  for (int it = 0; it < MAXIT; ++it) {
    k_thin<<<tg, tb, 0, stream>>>(bufA, bufB, 0, flags, gridBlocks);
    k_thin<<<tg, tb, 0, stream>>>(bufB, bufA, 1, flags, gridBlocks);
  }

  dim3 cg(WW / 16, HH / 8, NN);
  k_cmap<<<cg, 256, 0, stream>>>(bufA, Cpl);
  k_dmap<<<cg, 256, 0, stream>>>(Cpl, Dm);
  k_reduce<<<nhw / 256, 256, 0, stream>>>(logits, target, Dm, partials, nhw);
  k_final<<<1, 256, 0, stream>>>(partials, out, HWSZ / 256);
}
